// SensorClassifierDualEncoder_1589137899925
// MI455X (gfx1250) — matrix operand order repaired and checked by reading, not compiled, not run on hardware
//
#include <hip/hip_runtime.h>

// ---------------------------------------------------------------------------
// SensorClassifierDualEncoder for MI455X (gfx1250, wave32, WMMA bf16)
// ---------------------------------------------------------------------------

#define BB 32768
#define SS 5
#define DD 512
#define EE 256
#define HH 4
#define FF 2048
#define LL 2
#define CC 5
#define MM (BB * SS)   // 163840 token rows

typedef __attribute__((ext_vector_type(16))) __bf16 bf16x16;
typedef __attribute__((ext_vector_type(8)))  float  floatx8;

union BFrag {
    bf16x16 v;
    uint4   q[2];
    unsigned short u[16];
};

__device__ __forceinline__ unsigned short f32_to_bf16_rn(float f) {
    unsigned int u = __float_as_uint(f);
    unsigned int lsb = (u >> 16) & 1u;
    u += 0x7fffu + lsb;
    return (unsigned short)(u >> 16);
}

__device__ __forceinline__ float bf16_bits_to_f32(unsigned int bits16) {
    return __uint_as_float(bits16 << 16);
}

// ---------------------------------------------------------------------------
// f32 -> bf16 conversion, vectorized x4 (all arrays have size % 4 == 0)
// ---------------------------------------------------------------------------
__global__ __launch_bounds__(256)
void cvt_f32_bf16_kernel(const float4* __restrict__ in,
                         ushort4* __restrict__ out, int n4) {
    int i = blockIdx.x * blockDim.x + threadIdx.x;
    if (i < n4) {
        float4 f = in[i];
        ushort4 o;
        o.x = f32_to_bf16_rn(f.x);
        o.y = f32_to_bf16_rn(f.y);
        o.z = f32_to_bf16_rn(f.z);
        o.w = f32_to_bf16_rn(f.w);
        out[i] = o;
    }
}

// ---------------------------------------------------------------------------
// Generic WMMA GEMM: out[m,n] = act( A[m,:] . W[n,:] + bias[n] )
//   A: [M,K] bf16, W: [N,K] bf16, K%64==0, N%64==0, M%128==0
//   Wave tile: 32 rows x 64 cols (2x4 accumulators), 4 waves/block.
//   Two-stage register ping-pong pipeline over k (unroll by 64).
// ---------------------------------------------------------------------------
template<bool RELU>
__global__ __launch_bounds__(128)
void gemm_wmma_kernel(const unsigned short* __restrict__ A,
                      const unsigned short* __restrict__ W,
                      const float* __restrict__ bias,
                      float* __restrict__ outF,
                      unsigned short* __restrict__ outB,
                      int M, int N, int K) {
    const int lane = threadIdx.x & 31;
    const int wave = threadIdx.x >> 5;
    const int grp  = lane >> 4;     // 0: lanes 0-15, 1: lanes 16-31
    const int l16  = lane & 15;
    const int row0 = (blockIdx.x * 4 + wave) * 32;
    const int col0 = blockIdx.y * 64;
    if (row0 >= M) return;

    const floatx8 zero = {};
    floatx8 acc[2][4];
#pragma unroll
    for (int rb = 0; rb < 2; ++rb)
#pragma unroll
        for (int t = 0; t < 4; ++t) acc[rb][t] = zero;

    const unsigned short* arow0 = A + (size_t)(row0 + l16) * K;
    const unsigned short* arow1 = A + (size_t)(row0 + 16 + l16) * K;
    const unsigned short* wrow[4];
#pragma unroll
    for (int t = 0; t < 4; ++t)
        wrow[t] = W + (size_t)(col0 + t * 16 + l16) * K + grp * 16;

    BFrag a[2][2], b[2][4];   // [stage][frag]

    auto loadStage = [&](int s, int k) {
        // A fragment (16x32 bf16): lane row = l16; k = grp*16+h (contig 32B)
        // -> two consecutive 16B chunks per row block.
        a[s][0].q[0] = *(const uint4*)(arow0 + k + grp * 16);
        a[s][0].q[1] = *(const uint4*)(arow0 + k + grp * 16 + 8);
        a[s][1].q[0] = *(const uint4*)(arow1 + k + grp * 16);
        a[s][1].q[1] = *(const uint4*)(arow1 + k + grp * 16 + 8);
#pragma unroll
        for (int t = 0; t < 4; ++t) {
            // B fragment (32x16 bf16): lane col = l16; k = grp*16+h (contig 32B)
            b[s][t].q[0] = *(const uint4*)(wrow[t] + k);
            b[s][t].q[1] = *(const uint4*)(wrow[t] + k + 8);
        }
    };
    auto compute = [&](int s) {
#pragma unroll
        for (int t = 0; t < 4; ++t) {
            acc[0][t] = __builtin_amdgcn_wmma_f32_16x16x32_bf16(
                false, a[s][0].v, false, b[s][t].v, (short)0, acc[0][t], false, false);
            acc[1][t] = __builtin_amdgcn_wmma_f32_16x16x32_bf16(
                false, a[s][1].v, false, b[s][t].v, (short)0, acc[1][t], false, false);
        }
    };

    loadStage(0, 0);
    int k0 = 0;
    for (; k0 < K - 64; k0 += 64) {
        loadStage(1, k0 + 32);
        compute(0);
        loadStage(0, k0 + 64);
        compute(1);
    }
    loadStage(1, k0 + 32);
    compute(0);
    compute(1);

#pragma unroll
    for (int rb = 0; rb < 2; ++rb) {
#pragma unroll
        for (int t = 0; t < 4; ++t) {
            const int col = col0 + t * 16 + l16;
            const float bv = bias ? bias[col] : 0.0f;
            float vals[8];
#pragma unroll
            for (int r = 0; r < 8; ++r) {
                float v = acc[rb][t][r] + bv;
                if (RELU) v = fmaxf(v, 0.0f);
                vals[r] = v;
            }
            const int rbase = row0 + rb * 16 + grp * 8;
            if (outF) {
#pragma unroll
                for (int r = 0; r < 8; ++r)
                    outF[(size_t)(rbase + r) * N + col] = vals[r];
            }
            if (outB) {
#pragma unroll
                for (int r = 0; r < 8; ++r)
                    outB[(size_t)(rbase + r) * N + col] = f32_to_bf16_rn(vals[r]);
            }
        }
    }
}

// ---------------------------------------------------------------------------
// Embed: h = relu( select(sensor==0, x.Wg^T + bg, x.Wa^T + ba) )
// Dual-expert GEMM sharing A fragments. K=512, N=256.
// Wave tile: 32 rows x 32 cols x 2 experts, same ping-pong pipeline.
// ---------------------------------------------------------------------------
__global__ __launch_bounds__(128)
void embed_wmma_kernel(const unsigned short* __restrict__ xb,   // [M,512]
                       const unsigned short* __restrict__ wg,   // [256,512]
                       const unsigned short* __restrict__ wa,   // [256,512]
                       const float* __restrict__ gb,
                       const float* __restrict__ ab,
                       const int* __restrict__ stypes,          // [M]
                       float* __restrict__ h32,                 // [M,256]
                       unsigned short* __restrict__ hb) {       // [M,256]
    const int lane = threadIdx.x & 31;
    const int wave = threadIdx.x >> 5;
    const int grp  = lane >> 4;
    const int l16  = lane & 15;
    const int row0 = (blockIdx.x * 4 + wave) * 32;
    const int col0 = blockIdx.y * 32;

    const floatx8 zero = {};
    floatx8 accg[2][2], acca[2][2];
#pragma unroll
    for (int rb = 0; rb < 2; ++rb)
#pragma unroll
        for (int t = 0; t < 2; ++t) { accg[rb][t] = zero; acca[rb][t] = zero; }

    const unsigned short* arow0 = xb + (size_t)(row0 + l16) * DD;
    const unsigned short* arow1 = xb + (size_t)(row0 + 16 + l16) * DD;
    size_t wo[2];
#pragma unroll
    for (int t = 0; t < 2; ++t)
        wo[t] = (size_t)(col0 + t * 16 + l16) * DD + grp * 16;

    BFrag a[2][2], bg[2][2], ba[2][2];   // [stage][frag]

    auto loadStage = [&](int s, int k) {
        a[s][0].q[0] = *(const uint4*)(arow0 + k + grp * 16);
        a[s][0].q[1] = *(const uint4*)(arow0 + k + grp * 16 + 8);
        a[s][1].q[0] = *(const uint4*)(arow1 + k + grp * 16);
        a[s][1].q[1] = *(const uint4*)(arow1 + k + grp * 16 + 8);
#pragma unroll
        for (int t = 0; t < 2; ++t) {
            bg[s][t].q[0] = *(const uint4*)(wg + wo[t] + k);
            bg[s][t].q[1] = *(const uint4*)(wg + wo[t] + k + 8);
            ba[s][t].q[0] = *(const uint4*)(wa + wo[t] + k);
            ba[s][t].q[1] = *(const uint4*)(wa + wo[t] + k + 8);
        }
    };
    auto compute = [&](int s) {
#pragma unroll
        for (int rb = 0; rb < 2; ++rb)
#pragma unroll
            for (int t = 0; t < 2; ++t) {
                accg[rb][t] = __builtin_amdgcn_wmma_f32_16x16x32_bf16(
                    false, a[s][rb].v, false, bg[s][t].v, (short)0, accg[rb][t], false, false);
                acca[rb][t] = __builtin_amdgcn_wmma_f32_16x16x32_bf16(
                    false, a[s][rb].v, false, ba[s][t].v, (short)0, acca[rb][t], false, false);
            }
    };

    loadStage(0, 0);
    int k0 = 0;
    for (; k0 < DD - 64; k0 += 64) {
        loadStage(1, k0 + 32);
        compute(0);
        loadStage(0, k0 + 64);
        compute(1);
    }
    loadStage(1, k0 + 32);
    compute(0);
    compute(1);

#pragma unroll
    for (int rb = 0; rb < 2; ++rb) {
        const int rbase = row0 + rb * 16 + grp * 8;
        int st[8];
#pragma unroll
        for (int r = 0; r < 8; ++r) st[r] = stypes[rbase + r];
#pragma unroll
        for (int t = 0; t < 2; ++t) {
            const int col = col0 + t * 16 + l16;
            const float bgv = gb[col];
            const float bav = ab[col];
#pragma unroll
            for (int r = 0; r < 8; ++r) {
                float v = (st[r] == 0) ? (accg[rb][t][r] + bgv)
                                       : (acca[rb][t][r] + bav);
                v = fmaxf(v, 0.0f);
                const size_t idx = (size_t)(rbase + r) * EE + col;
                h32[idx] = v;
                hb[idx]  = f32_to_bf16_rn(v);
            }
        }
    }
}

// ---------------------------------------------------------------------------
// Attention, S=5, H=4, DH=64. One block (256 thr) per batch item.
// qkv: [B*S, 768] bf16 -> widened to f32 in LDS -> aob: [B*S, 256] bf16
// ---------------------------------------------------------------------------
__global__ __launch_bounds__(256)
void attention_kernel(const unsigned short* __restrict__ qkv,
                      unsigned short* __restrict__ aob) {
    __shared__ float sq[SS][EE];
    __shared__ float sk[SS][EE];
    __shared__ float sv[SS][EE];
    __shared__ float sattn[HH][SS][SS];

    const int b = blockIdx.x;
    const unsigned int* base = (const unsigned int*)(qkv + (size_t)b * SS * (3 * EE));

    // 5*768 bf16 = 1920 uint pairs; segment boundaries (multiples of 256) are
    // even so a pair never straddles q/k/v.
    for (int i = threadIdx.x; i < SS * 3 * EE / 2; i += 256) {
        const unsigned int u = base[i];
        const int e = i * 2;
        const int s = e / (3 * EE);
        const int c = e - s * (3 * EE);
        const float v0 = bf16_bits_to_f32(u & 0xffffu);
        const float v1 = bf16_bits_to_f32(u >> 16);
        if (c < EE)          { sq[s][c] = v0;          sq[s][c + 1] = v1; }
        else if (c < 2 * EE) { sk[s][c - EE] = v0;     sk[s][c - EE + 1] = v1; }
        else                 { sv[s][c - 2 * EE] = v0; sv[s][c - 2 * EE + 1] = v1; }
    }
    __syncthreads();

    if (threadIdx.x < HH * SS) {
        const int h = threadIdx.x / SS;
        const int s = threadIdx.x % SS;
        float sc[SS];
        float mx = -1e30f;
#pragma unroll
        for (int j = 0; j < SS; ++j) {
            float d = 0.0f;
            for (int dd = 0; dd < 64; ++dd)
                d += sq[s][h * 64 + dd] * sk[j][h * 64 + dd];
            sc[j] = d * 0.125f;  // 1/sqrt(64)
            mx = fmaxf(mx, sc[j]);
        }
        float se = 0.0f;
#pragma unroll
        for (int j = 0; j < SS; ++j) { sc[j] = __expf(sc[j] - mx); se += sc[j]; }
        const float inv = 1.0f / se;
#pragma unroll
        for (int j = 0; j < SS; ++j) sattn[h][s][j] = sc[j] * inv;
    }
    __syncthreads();

    const int h = threadIdx.x >> 6;
    const int d = threadIdx.x & 63;
#pragma unroll
    for (int s = 0; s < SS; ++s) {
        float o = 0.0f;
#pragma unroll
        for (int j = 0; j < SS; ++j) o += sattn[h][s][j] * sv[j][h * 64 + d];
        aob[((size_t)b * SS + s) * EE + h * 64 + d] = f32_to_bf16_rn(o);
    }
}

// ---------------------------------------------------------------------------
// residual add + LayerNorm over 256. One wave per row; in-place on h32.
// ---------------------------------------------------------------------------
__global__ __launch_bounds__(256)
void add_ln_kernel(float* __restrict__ h32,          // residual in/out [M,256]
                   const float* __restrict__ x,      // GEMM output     [M,256]
                   const float* __restrict__ gamma,
                   const float* __restrict__ beta,
                   unsigned short* __restrict__ hb) {// bf16 mirror out
    const int row  = blockIdx.x * 8 + (threadIdx.x >> 5);
    const int lane = threadIdx.x & 31;
    float* r = h32 + (size_t)row * EE;
    const float* xx = x + (size_t)row * EE;

    float v[8];
    float s = 0.0f;
#pragma unroll
    for (int i = 0; i < 8; ++i) {
        v[i] = r[lane * 8 + i] + xx[lane * 8 + i];
        s += v[i];
    }
#pragma unroll
    for (int off = 16; off > 0; off >>= 1) s += __shfl_xor(s, off, 32);
    const float mean = s * (1.0f / 256.0f);

    float vs = 0.0f;
#pragma unroll
    for (int i = 0; i < 8; ++i) { const float d = v[i] - mean; vs += d * d; }
#pragma unroll
    for (int off = 16; off > 0; off >>= 1) vs += __shfl_xor(vs, off, 32);
    const float rstd = rsqrtf(vs * (1.0f / 256.0f) + 1e-5f);

#pragma unroll
    for (int i = 0; i < 8; ++i) {
        const int c = lane * 8 + i;
        const float o = (v[i] - mean) * rstd * gamma[c] + beta[c];
        r[c] = o;
        hb[(size_t)row * EE + c] = f32_to_bf16_rn(o);
    }
}

// ---------------------------------------------------------------------------
// Final projection: out[b,c] = z2[b,:] . out_w[c,:] + out_b[c]   (C=5, K=64)
// ---------------------------------------------------------------------------
__global__ __launch_bounds__(256)
void head_out_kernel(const float* __restrict__ z2,   // [B,64]
                     const float* __restrict__ ow,   // [5,64]
                     const float* __restrict__ ob,
                     float* __restrict__ out) {      // [B,5]
    const int i = blockIdx.x * blockDim.x + threadIdx.x;
    if (i >= BB * CC) return;
    const int b = i / CC;
    const int c = i % CC;
    const float* z = z2 + (size_t)b * 64;
    float s = ob[c];
#pragma unroll
    for (int d = 0; d < 64; ++d) s += z[d] * ow[c * 64 + d];
    out[i] = s;
}

// ---------------------------------------------------------------------------
// Host-side orchestration
// ---------------------------------------------------------------------------
extern "C" void kernel_launch(void* const* d_in, const int* in_sizes, int n_in,
                              void* d_out, int out_size, void* d_ws, size_t ws_size,
                              hipStream_t stream) {
    const float* x        = (const float*)d_in[0];
    const int*   stypes   = (const int*)d_in[1];
    const float* gps_w    = (const float*)d_in[2];
    const float* gps_b    = (const float*)d_in[3];
    const float* ang_w    = (const float*)d_in[4];
    const float* ang_b    = (const float*)d_in[5];
    const float* qkv_w    = (const float*)d_in[6];
    const float* qkv_b    = (const float*)d_in[7];
    const float* attn_ow  = (const float*)d_in[8];
    const float* attn_ob  = (const float*)d_in[9];
    const float* ln1_g    = (const float*)d_in[10];
    const float* ln1_b    = (const float*)d_in[11];
    const float* ff1_w    = (const float*)d_in[12];
    const float* ff1_b    = (const float*)d_in[13];
    const float* ff2_w    = (const float*)d_in[14];
    const float* ff2_b    = (const float*)d_in[15];
    const float* ln2_g    = (const float*)d_in[16];
    const float* ln2_b    = (const float*)d_in[17];
    const float* fc1_w    = (const float*)d_in[18];
    const float* fc1_b    = (const float*)d_in[19];
    const float* fc2_w    = (const float*)d_in[20];
    const float* fc2_b    = (const float*)d_in[21];
    const float* out_w    = (const float*)d_in[22];
    const float* out_b    = (const float*)d_in[23];
    float* out = (float*)d_out;

    char* ws = (char*)d_ws;
    size_t off = 0;
    auto alloc = [&](size_t bytes) -> void* {
        off = (off + 255) & ~(size_t)255;
        void* p = ws + off;
        off += bytes;
        return p;
    };

    unsigned short* xb    = (unsigned short*)alloc((size_t)MM * DD * 2);
    unsigned short* hb    = (unsigned short*)alloc((size_t)MM * EE * 2);
    float*          h32   = (float*)alloc((size_t)MM * EE * 4);
    unsigned short* qkvb  = (unsigned short*)alloc((size_t)MM * 3 * EE * 2);
    unsigned short* aob   = (unsigned short*)alloc((size_t)MM * EE * 2);
    float*          tmp32 = (float*)alloc((size_t)MM * EE * 4);
    unsigned short* actf1 = (unsigned short*)alloc((size_t)MM * FF * 2);
    unsigned short* z1b   = (unsigned short*)alloc((size_t)BB * 128 * 2);
    float*          z2_32 = (float*)alloc((size_t)BB * 64 * 4);

    unsigned short* wg    = (unsigned short*)alloc((size_t)EE * DD * 2);
    unsigned short* wa    = (unsigned short*)alloc((size_t)EE * DD * 2);
    unsigned short* wqkv  = (unsigned short*)alloc((size_t)LL * 3 * EE * EE * 2);
    unsigned short* wao   = (unsigned short*)alloc((size_t)LL * EE * EE * 2);
    unsigned short* wf1   = (unsigned short*)alloc((size_t)LL * FF * EE * 2);
    unsigned short* wf2   = (unsigned short*)alloc((size_t)LL * EE * FF * 2);
    unsigned short* wfc1  = (unsigned short*)alloc((size_t)128 * (SS * EE) * 2);
    unsigned short* wfc2  = (unsigned short*)alloc((size_t)64 * 128 * 2);

    auto cvt = [&](const float* in, unsigned short* o, int n) {
        int n4 = n / 4;
        cvt_f32_bf16_kernel<<<(n4 + 255) / 256, 256, 0, stream>>>(
            (const float4*)in, (ushort4*)o, n4);
    };
    cvt(x,       xb,   MM * DD);
    cvt(gps_w,   wg,   EE * DD);
    cvt(ang_w,   wa,   EE * DD);
    cvt(qkv_w,   wqkv, LL * 3 * EE * EE);
    cvt(attn_ow, wao,  LL * EE * EE);
    cvt(ff1_w,   wf1,  LL * FF * EE);
    cvt(ff2_w,   wf2,  LL * EE * FF);
    cvt(fc1_w,   wfc1, 128 * SS * EE);
    cvt(fc2_w,   wfc2, 64 * 128);

    // Embed + dual-expert select (32x32x2 tiles)
    embed_wmma_kernel<<<dim3(MM / 128, EE / 32), 128, 0, stream>>>(
        xb, wg, wa, gps_b, ang_b, stypes, h32, hb);

    // Transformer layers
    for (int l = 0; l < LL; ++l) {
        gemm_wmma_kernel<false><<<dim3(MM / 128, (3 * EE) / 64), 128, 0, stream>>>(
            hb, wqkv + (size_t)l * 3 * EE * EE, qkv_b + (size_t)l * 3 * EE,
            nullptr, qkvb, MM, 3 * EE, EE);

        attention_kernel<<<BB, 256, 0, stream>>>(qkvb, aob);

        gemm_wmma_kernel<false><<<dim3(MM / 128, EE / 64), 128, 0, stream>>>(
            aob, wao + (size_t)l * EE * EE, attn_ob + (size_t)l * EE,
            tmp32, nullptr, MM, EE, EE);

        add_ln_kernel<<<MM / 8, 256, 0, stream>>>(
            h32, tmp32, ln1_g + (size_t)l * EE, ln1_b + (size_t)l * EE, hb);

        gemm_wmma_kernel<true><<<dim3(MM / 128, FF / 64), 128, 0, stream>>>(
            hb, wf1 + (size_t)l * FF * EE, ff1_b + (size_t)l * FF,
            nullptr, actf1, MM, FF, EE);

        gemm_wmma_kernel<false><<<dim3(MM / 128, EE / 64), 128, 0, stream>>>(
            actf1, wf2 + (size_t)l * EE * FF, ff2_b + (size_t)l * EE,
            tmp32, nullptr, MM, EE, FF);

        add_ln_kernel<<<MM / 8, 256, 0, stream>>>(
            h32, tmp32, ln2_g + (size_t)l * EE, ln2_b + (size_t)l * EE, hb);
    }

    // Head: hb is [B, S*E] bf16 contiguous by construction
    gemm_wmma_kernel<false><<<dim3(BB / 128, 128 / 64), 128, 0, stream>>>(
        hb, wfc1, fc1_b, nullptr, z1b, BB, 128, SS * EE);

    gemm_wmma_kernel<false><<<dim3(BB / 128, 64 / 64), 128, 0, stream>>>(
        z1b, wfc2, fc2_b, z2_32, nullptr, BB, 64, 128);

    head_out_kernel<<<(BB * CC + 255) / 256, 256, 0, stream>>>(
        z2_32, out_w, out_b, out);
}
